// DiffusionGenerator_17712445129054
// MI455X (gfx1250) — matrix operand order repaired and checked by reading, not compiled, not run on hardware
//
#include <hip/hip_runtime.h>
#include <hip/hip_bf16.h>
#include <math.h>

// ---------------- constants ----------------
constexpr int Bc = 4, Nc = 256, Hc = 256, Dc = 64, FFc = 1024;
constexpr int Mc = Bc * Nc;          // 1024 rows for all token-wise GEMMs
constexpr int Vc = 256, Lc = 128;

typedef __attribute__((ext_vector_type(16))) __bf16 v16bf;
typedef __attribute__((ext_vector_type(8)))  float  v8f;

// ---- Tensor Data Mover availability (arity differs per toolchain) ----
#if defined(__gfx1250__) && __has_builtin(__builtin_amdgcn_tensor_load_to_lds) && \
    __has_builtin(__builtin_amdgcn_s_wait_tensorcnt)
#define USE_TDM 1
typedef __attribute__((ext_vector_type(4))) unsigned int u32x4_t;
typedef __attribute__((ext_vector_type(8))) int          i32x8_t;
typedef __attribute__((ext_vector_type(4))) int          i32x4_t;
#if __clang_major__ >= 23
#define TDM_LOAD(g0, g1, g2, g3) \
  { i32x8_t __g4 = {0, 0, 0, 0, 0, 0, 0, 0}; \
    __builtin_amdgcn_tensor_load_to_lds((g0), (g1), (g2), (g3), __g4, 0); }
#else
#define TDM_LOAD(g0, g1, g2, g3) \
  __builtin_amdgcn_tensor_load_to_lds((g0), (g1), (g2), (g3), 0)
#endif
#endif

__device__ __forceinline__ float gelu_f(float x) {
  return 0.5f * x * (1.0f + erff(x * 0.70710678118654752f));
}

// ---------------- WMMA fragment loaders from global (used by attention) ----------------
// A (16x32 bf16): lanes 0-15 / 16-31 hold M = lane&15; frag elements 0..15 are
// K = half*16 .. half*16+15 (contiguous).
__device__ __forceinline__ v16bf load_a_frag(const float* __restrict__ A,
                                             int row, int lda, int kc, int half) {
  v16bf a;
  const float* p = A + (size_t)row * lda + kc;
#pragma unroll
  for (int e = 0; e < 8; ++e) {
    a[e]     = (__bf16)p[half * 16 + e];
    a[e + 8] = (__bf16)p[half * 16 + 8 + e];
  }
  return a;
}

// B (32x16 bf16), row-major K x N: lane holds column n = lane&15;
// lanes 0-15: K = kc..kc+15 sequential, lanes 16-31: K = kc+16..kc+31.
__device__ __forceinline__ v16bf load_b_frag(const float* __restrict__ W,
                                             int col, int ldb, int kc, int half) {
  v16bf b;
#pragma unroll
  for (int e = 0; e < 16; ++e) {
    int k = kc + half * 16 + e;
    b[e] = (__bf16)W[(size_t)k * ldb + col];
  }
  return b;
}

// B^T variant: B[k][n] = P[n*ld + k] (P row-major, rows are the N index).
__device__ __forceinline__ v16bf load_bT_frag(const float* __restrict__ P,
                                              int nrow, int ld, int kc, int half) {
  v16bf b;
  const float* p = P + (size_t)nrow * ld + kc + half * 16;
#pragma unroll
  for (int e = 0; e < 16; ++e) b[e] = (__bf16)p[e];
  return b;
}

// ---------------- tiled WMMA GEMM: out = ACT(A@W + bias [+ residual]) ----------------
// 128 threads = 4 waves per block; block tile 64x64; each wave owns 16 rows x 64 cols
// (4 accumulators, A-fragment reused 4x). A and B staged per 32-wide K-chunk into LDS
// as bf16; B transposed in LDS so every fragment is contiguous 16B ds loads.
constexpr int LDP = 40;   // padded LDS row stride in bf16 (80B: 16B-aligned, bank-spread)

template <int ACT, bool HAS_BIAS, bool HAS_RES>
__global__ __launch_bounds__(128) void wmma_gemm_tile_kernel(
    const float* __restrict__ A, const float* __restrict__ W,
    const float* __restrict__ bias, const float* __restrict__ residual,
    float* __restrict__ out, int M, int K, int Nout) {
  __shared__ __bf16 lsA[64 * LDP];   // [row][k]
  __shared__ __bf16 lsB[64 * LDP];   // [col][k]  (transposed)
  const int t = threadIdx.x;
  const int wv = t >> 5;
  const int lane = t & 31;
  const int half = lane >> 4, m = lane & 15;
  const int rowBase = blockIdx.y * 64, colBase = blockIdx.x * 64;

  v8f acc[4] = {};

  for (int kc = 0; kc < K; kc += 32) {
    if (kc) __syncthreads();
    // stage A: 64 rows x 32 K, coalesced float4 loads, bf16 into LDS
#pragma unroll
    for (int i = 0; i < 4; ++i) {
      int idx = t + i * 128;                 // 0..511 float4s
      int ar = idx >> 3, ac4 = idx & 7;
      const float4 v = *(const float4*)(A + (size_t)(rowBase + ar) * K + kc + ac4 * 4);
      __bf16* dst = &lsA[ar * LDP + ac4 * 4];
      dst[0] = (__bf16)v.x; dst[1] = (__bf16)v.y;
      dst[2] = (__bf16)v.z; dst[3] = (__bf16)v.w;
    }
    // stage B: 32 K-rows x 64 cols, coalesced float4 loads, transposed into LDS
#pragma unroll
    for (int i = 0; i < 4; ++i) {
      int idx = t + i * 128;
      int br = idx >> 4, bc4 = idx & 15;     // k-row, col/4
      const float4 v = *(const float4*)(W + (size_t)(kc + br) * Nout + colBase + bc4 * 4);
      lsB[(bc4 * 4 + 0) * LDP + br] = (__bf16)v.x;
      lsB[(bc4 * 4 + 1) * LDP + br] = (__bf16)v.y;
      lsB[(bc4 * 4 + 2) * LDP + br] = (__bf16)v.z;
      lsB[(bc4 * 4 + 3) * LDP + br] = (__bf16)v.w;
    }
    __syncthreads();

    // A fragment for this wave (one contiguous 32B run)
    const __bf16* pa = &lsA[(wv * 16 + m) * LDP + half * 16];
    v16bf afrag;
#pragma unroll
    for (int e = 0; e < 8; ++e) { afrag[e] = pa[e]; afrag[e + 8] = pa[e + 8]; }

#pragma unroll
    for (int nt = 0; nt < 4; ++nt) {
      const __bf16* pb = &lsB[(nt * 16 + m) * LDP + half * 16];
      v16bf bfrag;
#pragma unroll
      for (int e = 0; e < 16; ++e) bfrag[e] = pb[e];
      acc[nt] = __builtin_amdgcn_wmma_f32_16x16x32_bf16(false, afrag, false, bfrag,
                                                        (short)0, acc[nt], false, false);
    }
  }

#pragma unroll
  for (int nt = 0; nt < 4; ++nt) {
    const int ocol = colBase + nt * 16 + m;
    const float bv = HAS_BIAS ? bias[ocol] : 0.0f;
#pragma unroll
    for (int r = 0; r < 8; ++r) {
      const int orow = rowBase + wv * 16 + r + half * 8;
      float v = acc[nt][r] + bv;
      if (HAS_RES) v += residual[(size_t)orow * Nout + ocol];
      if (ACT == 1) v = fmaxf(v, 0.0f);
      if (ACT == 2) v = gelu_f(v);
      out[(size_t)orow * Nout + ocol] = v;
    }
  }
}

// ---------------- attention scores: S[bh,i,j] = (q.k)/8 ----------------
__global__ __launch_bounds__(32) void attn_score_kernel(
    const float* __restrict__ Qb, const float* __restrict__ Kb,
    float* __restrict__ S, int rowStride, float scale) {
  const int bh = blockIdx.z, b = bh >> 2, hh = bh & 3;
  const int lane = threadIdx.x & 31, half = lane >> 4, m = lane & 15;
  const float* Q = Qb + (size_t)(b * Nc) * rowStride + hh * Dc;
  const float* Km = Kb + (size_t)(b * Nc) * rowStride + hh * Dc;
  v8f acc = {};
#pragma unroll
  for (int kc = 0; kc < Dc; kc += 32) {
    v16bf a  = load_a_frag(Q, blockIdx.y * 16 + m, rowStride, kc, half);
    v16bf bb = load_bT_frag(Km, blockIdx.x * 16 + m, rowStride, kc, half);
    acc = __builtin_amdgcn_wmma_f32_16x16x32_bf16(false, a, false, bb,
                                                  (short)0, acc, false, false);
  }
#pragma unroll
  for (int r = 0; r < 8; ++r) {
    int i = blockIdx.y * 16 + r + half * 8;
    int j = blockIdx.x * 16 + m;
    S[((size_t)bh * Nc + i) * Nc + j] = acc[r] * scale;
  }
}

// ---------------- attention output: AO[b,i,h,d] = sum_j P[bh,i,j] * V[b,j,h,d] ----------------
__global__ __launch_bounds__(32) void attn_av_kernel(
    const float* __restrict__ P, const float* __restrict__ Vb,
    float* __restrict__ AO, int rowStride) {
  const int bh = blockIdx.z, b = bh >> 2, hh = bh & 3;
  const int lane = threadIdx.x & 31, half = lane >> 4, m = lane & 15;
  const float* A = P + (size_t)bh * Nc * Nc;
  const float* W = Vb + (size_t)(b * Nc) * rowStride + hh * Dc + blockIdx.x * 16;
  v8f acc = {};
  for (int kc = 0; kc < Nc; kc += 32) {
    v16bf a  = load_a_frag(A, blockIdx.y * 16 + m, Nc, kc, half);
    v16bf bb = load_b_frag(W, m, rowStride, kc, half);
    acc = __builtin_amdgcn_wmma_f32_16x16x32_bf16(false, a, false, bb,
                                                  (short)0, acc, false, false);
  }
#pragma unroll
  for (int r = 0; r < 8; ++r) {
    int i = blockIdx.y * 16 + r + half * 8;
    AO[((size_t)(b * Nc + i)) * Hc + hh * Dc + blockIdx.x * 16 + m] = acc[r];
  }
}

// ---------------- masked softmax over rows of S (B*HEADS, N, N) ----------------
__global__ __launch_bounds__(256) void masked_softmax_kernel(
    float* __restrict__ S, const float* __restrict__ nbr) {
  const int bh = blockIdx.y, b = bh >> 2;
  const int i = blockIdx.x, j = threadIdx.x;
  __shared__ float red[256];
  float* row = S + ((size_t)bh * Nc + i) * Nc;
  float mask = nbr ? nbr[((size_t)b * Nc + i) * Nc + j] : 1.0f;
  float v = row[j];
  float mv = (mask != 0.0f) ? v : -1e30f;
  red[j] = mv; __syncthreads();
  for (int s = 128; s > 0; s >>= 1) { if (j < s) red[j] = fmaxf(red[j], red[j + s]); __syncthreads(); }
  float mx = red[0]; __syncthreads();
  float e = __expf(mv - mx);
  red[j] = e; __syncthreads();
  for (int s = 128; s > 0; s >>= 1) { if (j < s) red[j] += red[j + s]; __syncthreads(); }
  float sum = red[0];
  row[j] = e / sum * mask;
}

// ---------------- layernorm over last dim (256) ----------------
__global__ __launch_bounds__(256) void layernorm_kernel(
    const float* __restrict__ in, float* __restrict__ out,
    const float* __restrict__ g, const float* __restrict__ bta) {
  const int rowi = blockIdx.x, j = threadIdx.x;
  __shared__ float red[256];
  float x = in[(size_t)rowi * Hc + j];
  red[j] = x; __syncthreads();
  for (int s = 128; s > 0; s >>= 1) { if (j < s) red[j] += red[j + s]; __syncthreads(); }
  float mean = red[0] * (1.0f / Hc); __syncthreads();
  float d = x - mean;
  red[j] = d * d; __syncthreads();
  for (int s = 128; s > 0; s >>= 1) { if (j < s) red[j] += red[j + s]; __syncthreads(); }
  float var = red[0] * (1.0f / Hc);
  out[(size_t)rowi * Hc + j] = d * rsqrtf(var + 1e-5f) * g[j] + bta[j];
}

// ---------------- edge logits (+ optional nbr mask, note the swapaxes) ----------------
// The hi-row tile (256 floats, contiguous) is staged into LDS via the Tensor Data
// Mover on gfx1250 (one TDM issue from wave 0; TDM ignores EXEC), synced with
// s_wait_tensorcnt; falls back to a cooperative load where TDM is unavailable.
__global__ __launch_bounds__(256) void edge_kernel(
    const float* __restrict__ HI, const float* __restrict__ HJ,
    const float* __restrict__ b1, const float* __restrict__ w2,
    const float* __restrict__ b2, float* __restrict__ edge_out,
    float* __restrict__ nbr_out) {
  const int b = blockIdx.y, i = blockIdx.x, j = threadIdx.x;
  __shared__ float s_hi[256];
  const float* hi_row = HI + ((size_t)(b * Nc + i)) * Hc;
#ifdef USE_TDM
  if (threadIdx.x < 32) {
    const unsigned long long ga = (unsigned long long)(uintptr_t)hi_row;
    const unsigned int lds = (unsigned int)(uintptr_t)(void*)s_hi;  // LDS aperture: low bits = LDS offset
    u32x4_t g0;
    g0[0] = 1u;                                               // count=1, user D#
    g0[1] = lds;                                              // lds_addr
    g0[2] = (unsigned int)ga;                                 // global_addr[31:0]
    g0[3] = (unsigned int)((ga >> 32) & 0x1FFFFFFu) | (2u << 30);  // addr[56:32] | type=2
    i32x8_t g1;
    g1[0] = (int)(2u << 16);                  // data_size=4B; no multicast/pad/iterate
    g1[1] = (int)((Hc & 0xFFFF) << 16);       // tensor_dim0[15:0]  (bits 63:48)
    g1[2] = (int)(((Hc >> 16) & 0xFFFF) | (1 << 16)); // tensor_dim0[31:16], tensor_dim1=1
    g1[3] = (int)((0) | (Hc << 16));          // tensor_dim1[31:16]=0, tile_dim0=256
    g1[4] = 1;                                // tile_dim1=1, tile_dim2=0
    g1[5] = Hc;                               // tensor_dim0_stride[31:0]
    g1[6] = 0;                                // stride0[47:32], stride1[15:0]
    g1[7] = 0;                                // stride1 hi
    i32x4_t gz = {0, 0, 0, 0};
    TDM_LOAD(g0, g1, gz, gz);
    __builtin_amdgcn_s_wait_tensorcnt(0);
  }
  __syncthreads();
  s_hi[j] += b1[j];
  __syncthreads();
#else
  s_hi[j] = hi_row[j] + b1[j];
  __syncthreads();
#endif
  const float* hj = HJ + ((size_t)(b * Nc + j)) * Hc;
  float acc = 0.0f;
  for (int k = 0; k < Hc; ++k) acc += gelu_f(s_hi[k] + hj[k]) * w2[k];
  acc += b2[0];
  if (edge_out) edge_out[((size_t)b * Nc + i) * Nc + j] = acc;
  if (nbr_out) {
    float sig = 1.0f / (1.0f + __expf(-acc));
    // nbr = swapaxes(sigmoid(el0)>0.3, 1, 2):  nbr[b][j][i] = f(el0[b][i][j])
    nbr_out[((size_t)b * Nc + j) * Nc + i] = (sig > 0.3f) ? 1.0f : 0.0f;
  }
}

// ---------------- small helpers ----------------
__global__ void naive_gemm_kernel(const float* __restrict__ A, const float* __restrict__ W,
                                  const float* __restrict__ bias, float* __restrict__ out,
                                  int M, int K, int Nout, int act) {
  int idx = blockIdx.x * blockDim.x + threadIdx.x;
  if (idx >= M * Nout) return;
  int r = idx / Nout, c = idx % Nout;
  float acc = bias ? bias[c] : 0.0f;
  for (int k = 0; k < K; ++k) acc += A[(size_t)r * K + k] * W[(size_t)k * Nout + c];
  if (act == 1)      acc = fmaxf(acc, 0.0f);
  else if (act == 2) acc = gelu_f(acc);
  out[idx] = acc;
}

__global__ __launch_bounds__(256) void time_embed_kernel(
    const float* __restrict__ time_tab, const int* __restrict__ t,
    const float* __restrict__ tw, const float* __restrict__ tb,
    float* __restrict__ out) {
  const int b = blockIdx.x, j = threadIdx.x;
  const float* row = time_tab + (size_t)t[b] * Hc;
  float acc = tb[j];
  for (int k = 0; k < Hc; ++k) acc += row[k] * tw[(size_t)k * Hc + j];
  out[b * Hc + j] = gelu_f(acc);
}

__global__ __launch_bounds__(256) void hinit_kernel(
    const int* __restrict__ tok, const float* __restrict__ tok_emb,
    const float* __restrict__ time_e, const float* __restrict__ pos,
    const float* __restrict__ lat, float* __restrict__ X) {
  int idx = blockIdx.x * 256 + threadIdx.x;          // B*N*H = 262144
  int c = idx & 255, bn = idx >> 8, b = bn >> 8, n = bn & 255;
  X[idx] = tok_emb[(size_t)tok[bn] * Hc + c] + time_e[b * Hc + c] +
           pos[(size_t)n * Hc + c] + lat[b * Hc + c];
}

__global__ void add_kernel(const float* __restrict__ a, const float* __restrict__ b,
                           float* __restrict__ o, int n) {
  int i = blockIdx.x * blockDim.x + threadIdx.x;
  if (i < n) o[i] = a[i] + b[i];
}

// ---------------- host-side GEMM dispatch (compile-time act/bias/res variants) ----------------
static inline void LGEMM(hipStream_t s, const float* A, const float* W, const float* bias,
                         const float* res, float* out, int M, int K, int Nout, int act) {
  dim3 g(Nout / 64, M / 64), blk(128);
  if (res) {
    wmma_gemm_tile_kernel<0, true, true><<<g, blk, 0, s>>>(A, W, bias, res, out, M, K, Nout);
  } else if (!bias) {
    wmma_gemm_tile_kernel<0, false, false><<<g, blk, 0, s>>>(A, W, bias, res, out, M, K, Nout);
  } else if (act == 0) {
    wmma_gemm_tile_kernel<0, true, false><<<g, blk, 0, s>>>(A, W, bias, res, out, M, K, Nout);
  } else if (act == 1) {
    wmma_gemm_tile_kernel<1, true, false><<<g, blk, 0, s>>>(A, W, bias, res, out, M, K, Nout);
  } else {
    wmma_gemm_tile_kernel<2, true, false><<<g, blk, 0, s>>>(A, W, bias, res, out, M, K, Nout);
  }
}

extern "C" void kernel_launch(void* const* d_in, const int* in_sizes, int n_in,
                              void* d_out, int out_size, void* d_ws, size_t ws_size,
                              hipStream_t stream) {
  (void)in_sizes; (void)n_in; (void)out_size; (void)ws_size;
  int a = 0;
  const float* z        = (const float*)d_in[a++];
  const int*   tok      = (const int*)  d_in[a++];
  const int*   tt       = (const int*)  d_in[a++];
  const float* lp1_w    = (const float*)d_in[a++];
  const float* lp1_b    = (const float*)d_in[a++];
  const float* lp2_w    = (const float*)d_in[a++];
  const float* lp2_b    = (const float*)d_in[a++];
  const float* tok_emb  = (const float*)d_in[a++];
  const float* time_tab = (const float*)d_in[a++];
  const float* time_w   = (const float*)d_in[a++];
  const float* time_b   = (const float*)d_in[a++];
  const float* pos_tab  = (const float*)d_in[a++];
  const float* enc_qkv_w= (const float*)d_in[a++];
  const float* enc_qkv_b= (const float*)d_in[a++];
  const float* enc_o_w  = (const float*)d_in[a++];
  const float* enc_o_b  = (const float*)d_in[a++];
  const float* enc_ln1_g= (const float*)d_in[a++];
  const float* enc_ln1_b= (const float*)d_in[a++];
  const float* enc_f1_w = (const float*)d_in[a++];
  const float* enc_f1_b = (const float*)d_in[a++];
  const float* enc_f2_w = (const float*)d_in[a++];
  const float* enc_f2_b = (const float*)d_in[a++];
  const float* enc_ln2_g= (const float*)d_in[a++];
  const float* enc_ln2_b= (const float*)d_in[a++];
  const float* gq_w     = (const float*)d_in[a++];
  const float* gq_b     = (const float*)d_in[a++];
  const float* gk_w     = (const float*)d_in[a++];
  const float* gk_b     = (const float*)d_in[a++];
  const float* gv_w     = (const float*)d_in[a++];
  const float* gv_b     = (const float*)d_in[a++];
  const float* gs_w     = (const float*)d_in[a++];
  const float* gs_b     = (const float*)d_in[a++];
  const float* g_ln1_g  = (const float*)d_in[a++];
  const float* g_ln1_b  = (const float*)d_in[a++];
  const float* g_f1_w   = (const float*)d_in[a++];
  const float* g_f1_b   = (const float*)d_in[a++];
  const float* g_f2_w   = (const float*)d_in[a++];
  const float* g_f2_b   = (const float*)d_in[a++];
  const float* g_ln2_g  = (const float*)d_in[a++];
  const float* g_ln2_b  = (const float*)d_in[a++];
  const float* nh1_w    = (const float*)d_in[a++];
  const float* nh1_b    = (const float*)d_in[a++];
  const float* nh2_w    = (const float*)d_in[a++];
  const float* nh2_b    = (const float*)d_in[a++];
  const float* eh1_w    = (const float*)d_in[a++];
  const float* eh1_b    = (const float*)d_in[a++];
  const float* eh2_w    = (const float*)d_in[a++];
  const float* eh2_b    = (const float*)d_in[a++];
  const float* sh1_w    = (const float*)d_in[a++];
  const float* sh1_b    = (const float*)d_in[a++];
  const float* sh2_w    = (const float*)d_in[a++];
  const float* sh2_b    = (const float*)d_in[a++];

  // workspace layout (floats)
  float* w    = (float*)d_ws;
  float* X    = w;                 // 1024x256 token states
  float* T1   = X    + 262144;     // 1024x256 scratch
  float* Qb   = T1   + 262144;     // 1024x256 (encoder: contiguous 1024x768 QKV spans Qb..Vb)
  float* Kb   = Qb   + 262144;
  float* Vb   = Kb   + 262144;
  float* Sb   = Vb   + 262144;     // 16 x 256 x 256 attention scores/probs
  float* AO   = Sb   + 1048576;    // 1024x256 attention output
  float* FFNb = AO   + 262144;     // 1024x1024
  float* HIb  = FFNb + 1048576;    // 1024x256
  float* HJb  = HIb  + 262144;     // 1024x256
  float* NBR  = HJb  + 262144;     // 4x256x256 float mask
  float* LAT  = NBR  + 262144;     // 4x256
  float* TS   = LAT  + 1024;       // 4x256 latent hidden
  float* TIMEe= TS   + 1024;       // 4x256 time embedding
  float* S1   = TIMEe+ 1024;       // 4x128 size-head hidden

  float* node_out = (float*)d_out;           // (4,256,256)
  float* edge_out = node_out + 262144;       // (4,256,256)
  float* size_out = edge_out + 262144;       // (4,256)

  // latent path + time embedding + h init
  naive_gemm_kernel<<<dim3(4), 256, 0, stream>>>(z, lp1_w, lp1_b, TS, Bc, Lc, Hc, 1);
  naive_gemm_kernel<<<dim3(4), 256, 0, stream>>>(TS, lp2_w, lp2_b, LAT, Bc, Hc, Hc, 0);
  time_embed_kernel<<<dim3(Bc), 256, 0, stream>>>(time_tab, tt, time_w, time_b, TIMEe);
  hinit_kernel<<<dim3(1024), 256, 0, stream>>>(tok, tok_emb, TIMEe, pos_tab, LAT, X);

  // size head (depends only on LAT)
  naive_gemm_kernel<<<dim3(2), 256, 0, stream>>>(LAT, sh1_w, sh1_b, S1, Bc, Hc, Hc / 2, 1);
  naive_gemm_kernel<<<dim3(4), 256, 0, stream>>>(S1, sh2_w, sh2_b, size_out, Bc, Hc / 2, Nc, 0);

  // ---------------- 2 encoder layers ----------------
  for (int l = 0; l < 2; ++l) {
    LGEMM(stream, X, enc_qkv_w + (size_t)l * Hc * 3 * Hc, enc_qkv_b + l * 3 * Hc,
          nullptr, Qb, Mc, Hc, 3 * Hc, 0);                           // fused QKV, stride 768
    attn_score_kernel<<<dim3(16, 16, 16), 32, 0, stream>>>(Qb, Qb + Hc, Sb, 3 * Hc, 0.125f);
    masked_softmax_kernel<<<dim3(Nc, 16), 256, 0, stream>>>(Sb, nullptr);
    attn_av_kernel<<<dim3(4, 16, 16), 32, 0, stream>>>(Sb, Qb + 2 * Hc, AO, 3 * Hc);
    LGEMM(stream, AO, enc_o_w + (size_t)l * Hc * Hc, enc_o_b + l * Hc, X, T1, Mc, Hc, Hc, 0);
    layernorm_kernel<<<dim3(Mc), 256, 0, stream>>>(T1, X, enc_ln1_g + l * Hc, enc_ln1_b + l * Hc);
    LGEMM(stream, X, enc_f1_w + (size_t)l * Hc * FFc, enc_f1_b + l * FFc, nullptr, FFNb, Mc, Hc, FFc, 1);
    LGEMM(stream, FFNb, enc_f2_w + (size_t)l * FFc * Hc, enc_f2_b + l * Hc, X, T1, Mc, FFc, Hc, 0);
    layernorm_kernel<<<dim3(Mc), 256, 0, stream>>>(T1, X, enc_ln2_g + l * Hc, enc_ln2_b + l * Hc);
  }

  // ---------------- neighbor mask from edge logits on h ----------------
  LGEMM(stream, X, eh1_w, nullptr, nullptr, HIb, Mc, Hc, Hc, 0);
  LGEMM(stream, X, eh1_w + (size_t)Hc * Hc, nullptr, nullptr, HJb, Mc, Hc, Hc, 0);
  edge_kernel<<<dim3(Nc, Bc), 256, 0, stream>>>(HIb, HJb, eh1_b, eh2_w, eh2_b, nullptr, NBR);

  // ---------------- 4 GNN blocks ----------------
  for (int l = 0; l < 4; ++l) {
    LGEMM(stream, X, gq_w + (size_t)l * Hc * Hc, gq_b + l * Hc, nullptr, Qb, Mc, Hc, Hc, 0);
    LGEMM(stream, X, gk_w + (size_t)l * Hc * Hc, gk_b + l * Hc, nullptr, Kb, Mc, Hc, Hc, 0);
    LGEMM(stream, X, gv_w + (size_t)l * Hc * Hc, gv_b + l * Hc, nullptr, Vb, Mc, Hc, Hc, 0);
    attn_score_kernel<<<dim3(16, 16, 16), 32, 0, stream>>>(Qb, Kb, Sb, Hc, 0.125f);
    masked_softmax_kernel<<<dim3(Nc, 16), 256, 0, stream>>>(Sb, NBR);
    attn_av_kernel<<<dim3(4, 16, 16), 32, 0, stream>>>(Sb, Vb, AO, Hc);
    add_kernel<<<dim3(1024), 256, 0, stream>>>(X, AO, T1, 262144);   // x + agg
    LGEMM(stream, X, gs_w + (size_t)l * Hc * Hc, gs_b + l * Hc, T1, T1, Mc, Hc, Hc, 0); // + x@sw+sb
    layernorm_kernel<<<dim3(Mc), 256, 0, stream>>>(T1, X, g_ln1_g + l * Hc, g_ln1_b + l * Hc);
    LGEMM(stream, X, g_f1_w + (size_t)l * Hc * FFc, g_f1_b + l * FFc, nullptr, FFNb, Mc, Hc, FFc, 2);
    LGEMM(stream, FFNb, g_f2_w + (size_t)l * FFc * Hc, g_f2_b + l * Hc, X, T1, Mc, FFc, Hc, 0);
    layernorm_kernel<<<dim3(Mc), 256, 0, stream>>>(T1, X, g_ln2_g + l * Hc, g_ln2_b + l * Hc);
  }

  // ---------------- heads ----------------
  LGEMM(stream, X, nh1_w, nh1_b, nullptr, T1, Mc, Hc, Hc, 2);
  LGEMM(stream, T1, nh2_w, nh2_b, nullptr, node_out, Mc, Hc, Vc, 0);
  LGEMM(stream, X, eh1_w, nullptr, nullptr, HIb, Mc, Hc, Hc, 0);
  LGEMM(stream, X, eh1_w + (size_t)Hc * Hc, nullptr, nullptr, HJb, Mc, Hc, Hc, 0);
  edge_kernel<<<dim3(Nc, Bc), 256, 0, stream>>>(HIb, HJb, eh1_b, eh2_w, eh2_b, edge_out, nullptr);
}
